// MultiHeadAttentionLayer_13434657702056
// MI455X (gfx1250) — compile-verified, matrix operand order checked by reading, not run on hardware
//
#include <hip/hip_runtime.h>

// ---------------------------------------------------------------------------
// Swin-style windowed cross attention for MI455X (gfx1250, wave32, WMMA).
// Pipeline: prep_weights -> proj_q -> proj_kv -> attn -> proj_o
// All intermediates bf16 in d_ws (~252 MB). Memory-bound problem: bf16
// halves intermediate traffic; rolls/partitions folded into store indexing.
// ---------------------------------------------------------------------------

typedef __attribute__((ext_vector_type(16))) __bf16 v16bf;
typedef __attribute__((ext_vector_type(8)))  __bf16 v8bf;
typedef __attribute__((ext_vector_type(8)))  float  v8f;
typedef __attribute__((ext_vector_type(4)))  float  v4f;

#define DEV static __device__ __forceinline__

// D = A*B + C, bf16 inputs, f32 accumulate.  (neg_a, A, neg_b, B, c_mod, C, reuse_a, reuse_b)
DEV v8f wmma_bf16(v16bf a, v16bf b, v8f c) {
  return __builtin_amdgcn_wmma_f32_16x16x32_bf16(false, a, false, b, (short)0, c, false, false);
}

// 16-bit A/B fragment load (ISA 7.12.2 16-bit A 16x32 layout):
// lane<16 holds K {koff..koff+7, koff+16..koff+23} of its row with koff=0,
// lane>=16 with koff=8.  Two contiguous 8-element (128-bit) loads.
DEV v16bf ldfrag_bf16(const __bf16* rowptr, int koff) {
  v8bf lo = *(const v8bf*)(rowptr + koff);
  v8bf hi = *(const v8bf*)(rowptr + koff + 16);
  return __builtin_shufflevector(lo, hi, 0,1,2,3,4,5,6,7,8,9,10,11,12,13,14,15);
}

// Same fragment, but sourced from fp32 memory (converted to bf16 on the fly).
DEV v16bf ldfrag_f32(const float* rowptr, int koff) {
  v4f a = *(const v4f*)(rowptr + koff);
  v4f b = *(const v4f*)(rowptr + koff + 4);
  v4f c = *(const v4f*)(rowptr + koff + 16);
  v4f d = *(const v4f*)(rowptr + koff + 20);
  v16bf r;
#pragma unroll
  for (int i = 0; i < 4; ++i) {
    r[i]      = (__bf16)a[i];
    r[i + 4]  = (__bf16)b[i];
    r[i + 8]  = (__bf16)c[i];
    r[i + 12] = (__bf16)d[i];
  }
  return r;
}

// ---------------------------------------------------------------------------
// Workspace layout (element offsets, __bf16)
// ---------------------------------------------------------------------------
static constexpr size_t N_Q   = 4ull * 6 * 16 * 16 * 256 * 32;  // 50,331,648
static constexpr size_t N_KV  = 4ull * 6 * 16 * 16 * 64 * 32;   // 12,582,912
static constexpr size_t N_O   = 4ull * 256 * 256 * 192;         // 50,331,648
static constexpr size_t OFF_Q    = 0;
static constexpr size_t OFF_K    = OFF_Q + N_Q;
static constexpr size_t OFF_V    = OFF_K + N_KV;
static constexpr size_t OFF_O    = OFF_V + N_KV;
static constexpr size_t OFF_WQT  = OFF_O + N_O;
static constexpr size_t OFF_WKVT = OFF_WQT + 192 * 192;
static constexpr size_t OFF_WOT  = OFF_WKVT + 192 * 384;

// ---------------------------------------------------------------------------
// Kernel 0: transpose + convert weights to bf16 (B-fragments then load as
// contiguous 128-bit chunks from W^T rows).
// ---------------------------------------------------------------------------
__global__ void prep_weights(const float* __restrict__ Wq,
                             const float* __restrict__ Wkv,
                             const float* __restrict__ Wo,
                             __bf16* __restrict__ wqT,
                             __bf16* __restrict__ wkvT,
                             __bf16* __restrict__ woT) {
  int i = blockIdx.x * 256 + threadIdx.x;  // 147456 total
  if (i < 36864) {
    int n = i / 192, k = i % 192;
    wqT[i] = (__bf16)Wq[k * 192 + n];
  } else if (i < 36864 + 73728) {
    int j = i - 36864;
    int n = j / 192, k = j % 192;          // n in [0,384)
    wkvT[j] = (__bf16)Wkv[k * 384 + n];
  } else if (i < 147456) {
    int j = i - 110592;
    int n = j / 192, k = j % 192;
    woT[j] = (__bf16)Wo[k * 192 + n];
  }
}

// ---------------------------------------------------------------------------
// Kernel 1: Q projection. 2048 blocks x 256 thr; wave = 16 rows x 192 cols.
// Stores bf16 into rolled(-8) + 16x16-window-partitioned layout:
//   qbuf[b][h][wy][wx][q(256)][dk(32)]
// ---------------------------------------------------------------------------
__global__ __launch_bounds__(256) void proj_q_kernel(const float* __restrict__ x,
                                                     const __bf16* __restrict__ wqT,
                                                     const float* __restrict__ bq,
                                                     __bf16* __restrict__ qbuf) {
  const int lane  = threadIdx.x & 31;
  const int wave  = threadIdx.x >> 5;
  const int koff  = (lane < 16) ? 0 : 8;
  const int nlo   = lane & 15;
  const int rhalf = (lane >> 4) << 3;
  const int rowbase = blockIdx.x * 128 + wave * 16;
  const int rowA = rowbase + nlo;

  v8f acc[12] = {};
#pragma unroll
  for (int kk = 0; kk < 6; ++kk) {
    v16bf af = ldfrag_f32(x + (size_t)rowA * 192 + kk * 32, koff);
#pragma unroll
    for (int ct = 0; ct < 12; ++ct) {
      v16bf bf = ldfrag_bf16(wqT + (size_t)(ct * 16 + nlo) * 192 + kk * 32, koff);
      acc[ct] = wmma_bf16(af, bf, acc[ct]);
    }
  }
#pragma unroll
  for (int ct = 0; ct < 12; ++ct) {
    const int c  = ct * 16 + nlo;
    const float bias = bq[c];
    const int h  = c >> 5, dk = c & 31;
#pragma unroll
    for (int r = 0; r < 8; ++r) {
      int p  = rowbase + r + rhalf;
      int b  = p >> 16;
      int y  = (p >> 8) & 255;
      int xx = p & 255;
      int y2 = (y + 248) & 255;   // roll -8
      int x2 = (xx + 248) & 255;
      int wy = y2 >> 4, iy = y2 & 15;
      int wx = x2 >> 4, ix = x2 & 15;
      size_t idx = (((size_t)((b * 6 + h) * 16 + wy) * 16 + wx) * 256 + (iy * 16 + ix)) * 32 + dk;
      qbuf[idx] = (__bf16)(acc[ct][r] + bias);
    }
  }
}

// ---------------------------------------------------------------------------
// Kernel 2: KV projection. 1024 blocks; wave = 16 rows x 192 cols (half of 384).
// Stores rolled(-4) + 8x8-window-partitioned: {k,v}buf[b][h][wy][wx][key(64)][dk(32)]
// ---------------------------------------------------------------------------
__global__ __launch_bounds__(256) void proj_kv_kernel(const float* __restrict__ z,
                                                      const __bf16* __restrict__ wkvT,
                                                      const float* __restrict__ bkv,
                                                      __bf16* __restrict__ kbuf,
                                                      __bf16* __restrict__ vbuf) {
  const int lane  = threadIdx.x & 31;
  const int wave  = threadIdx.x >> 5;
  const int koff  = (lane < 16) ? 0 : 8;
  const int nlo   = lane & 15;
  const int rhalf = (lane >> 4) << 3;
  const int rowtile = wave >> 1;
  const int colhalf = wave & 1;
  const int rowbase = blockIdx.x * 64 + rowtile * 16;
  const int rowA = rowbase + nlo;

  v8f acc[12] = {};
#pragma unroll
  for (int kk = 0; kk < 6; ++kk) {
    v16bf af = ldfrag_f32(z + (size_t)rowA * 192 + kk * 32, koff);
#pragma unroll
    for (int ct = 0; ct < 12; ++ct) {
      int n = colhalf * 192 + ct * 16 + nlo;
      v16bf bf = ldfrag_bf16(wkvT + (size_t)n * 192 + kk * 32, koff);
      acc[ct] = wmma_bf16(af, bf, acc[ct]);
    }
  }
#pragma unroll
  for (int ct = 0; ct < 12; ++ct) {
    const int c  = colhalf * 192 + ct * 16 + nlo;
    const float bias = bkv[c];
    const int kv = (c >= 192);
    const int cc = c - kv * 192;
    const int h  = cc >> 5, dk = cc & 31;
    __bf16* dst = kv ? vbuf : kbuf;
#pragma unroll
    for (int r = 0; r < 8; ++r) {
      int p  = rowbase + r + rhalf;
      int b  = p >> 14;
      int y  = (p >> 7) & 127;
      int xx = p & 127;
      int y2 = (y + 124) & 127;   // roll -4
      int x2 = (xx + 124) & 127;
      int wy = y2 >> 3, iy = y2 & 7;
      int wx = x2 >> 3, ix = x2 & 7;
      size_t idx = (((size_t)((b * 6 + h) * 16 + wy) * 16 + wx) * 64 + (iy * 8 + ix)) * 32 + dk;
      dst[idx] = (__bf16)(acc[ct][r] + bias);
    }
  }
}

// ---------------------------------------------------------------------------
// Kernel 3: windowed attention. One block per (b,h,wy,wx); 8 waves, 2 query
// tiles each.  dk=32 -> single WMMA per 16x16 score tile.  Softmax in
// accumulator layout via shfl_xor (16-lane groups match C-layout rows);
// P restaged through LDS for C->A layout; V^T staged in LDS for B-frags.
// Output written rolled-back/merged: obuf[b][y][x][h*32+dk] (bf16).
// ---------------------------------------------------------------------------
__global__ __launch_bounds__(256) void attn_kernel(const __bf16* __restrict__ qbuf,
                                                   const __bf16* __restrict__ kbuf,
                                                   const __bf16* __restrict__ vbuf,
                                                   const unsigned char* __restrict__ mask,
                                                   const float* __restrict__ rel_table,
                                                   __bf16* __restrict__ obuf) {
  __shared__ __bf16 vt[32][64];        // V^T : [dk][key]
  __shared__ __bf16 patt[8][16][64];   // per-wave P staging

  const int tid  = threadIdx.x;
  const int lane = tid & 31;
  const int wave = tid >> 5;
  const int bid  = blockIdx.x;                 // ((b*6+h)*16+wy)*16+wx
  const int wx = bid & 15, wy = (bid >> 4) & 15;
  const int bh = bid >> 8;
  const int h  = bh % 6, b = bh / 6;

  const __bf16* qwin = qbuf + (size_t)bid * 8192;
  const __bf16* kwin = kbuf + (size_t)bid * 2048;
  const __bf16* vwin = vbuf + (size_t)bid * 2048;
  const unsigned char* mwin = mask + (size_t)((h * 16 + wy) * 16 + wx) * 16384;

  // stage V^T
  {
    int key = tid >> 2;
    int dk0 = (tid & 3) * 8;
    v8bf vv = *(const v8bf*)(vwin + key * 32 + dk0);
#pragma unroll
    for (int j = 0; j < 8; ++j) vt[dk0 + j][key] = vv[j];
  }
  __syncthreads();

  const int koff  = (lane < 16) ? 0 : 8;
  const int nlo   = lane & 15;
  const int rhalf = (lane >> 4) << 3;
  const float SCALE = 0.17677669529663687f;   // 1/sqrt(32)

  for (int qi = 0; qi < 2; ++qi) {
    const int qt = wave * 2 + qi;

    // S = Q K^T  (one WMMA per key tile, K=dk=32)
    v16bf aq = ldfrag_bf16(qwin + (qt * 16 + nlo) * 32, koff);
    v8f att[4];
#pragma unroll
    for (int kt = 0; kt < 4; ++kt) {
      v16bf bk = ldfrag_bf16(kwin + (kt * 16 + nlo) * 32, koff);
      v8f zero = {};
      att[kt] = wmma_bf16(aq, bk, zero);
    }

    // scale + relative-position bias + mask (each (q,k) used exactly once)
#pragma unroll
    for (int kt = 0; kt < 4; ++kt) {
      const int k  = kt * 16 + nlo;
      const int ky = (k >> 3) * 2, kx2 = (k & 7) * 2;
#pragma unroll
      for (int r = 0; r < 8; ++r) {
        int q  = qt * 16 + r + rhalf;
        int qy = q >> 4, qx = q & 15;
        int ridx = (qy - ky + 14) * 30 + (qx - kx2 + 14);
        float a = att[kt][r] * SCALE + rel_table[ridx * 6 + h];
        if (mwin[q * 64 + k]) a = -1e9f;
        att[kt][r] = a;
      }
    }

    // row softmax across 64 keys (4 reg tiles x 16 lanes)
    float rs[8];
#pragma unroll
    for (int r = 0; r < 8; ++r) {
      float m = fmaxf(fmaxf(att[0][r], att[1][r]), fmaxf(att[2][r], att[3][r]));
#pragma unroll
      for (int off = 1; off < 16; off <<= 1) m = fmaxf(m, __shfl_xor(m, off, 32));
      float s = 0.f;
#pragma unroll
      for (int t = 0; t < 4; ++t) {
        float e = __expf(att[t][r] - m);
        att[t][r] = e;
        s += e;
      }
#pragma unroll
      for (int off = 1; off < 16; off <<= 1) s += __shfl_xor(s, off, 32);
      rs[r] = __builtin_amdgcn_rcpf(s);      // defer 1/sum to after P*V
    }

    // stash P (bf16) to LDS: C-layout -> A-layout transpose
#pragma unroll
    for (int kt = 0; kt < 4; ++kt)
#pragma unroll
      for (int r = 0; r < 8; ++r)
        patt[wave][r + rhalf][kt * 16 + nlo] = (__bf16)att[kt][r];
    asm volatile("s_wait_dscnt 0" ::: "memory");

    v16bf pa0 = ldfrag_bf16(&patt[wave][nlo][0], koff);   // keys 0..31
    v16bf pa1 = ldfrag_bf16(&patt[wave][nlo][32], koff);  // keys 32..63

#pragma unroll
    for (int n = 0; n < 2; ++n) {                          // dk halves
      v16bf vb0 = ldfrag_bf16(&vt[n * 16 + nlo][0], koff);
      v16bf vb1 = ldfrag_bf16(&vt[n * 16 + nlo][32], koff);
      v8f o = {};
      o = wmma_bf16(pa0, vb0, o);
      o = wmma_bf16(pa1, vb1, o);
#pragma unroll
      for (int r = 0; r < 8; ++r) {
        int q  = qt * 16 + r + rhalf;
        int qy = q >> 4, qx = q & 15;
        int y  = (wy * 16 + qy + 8) & 255;   // roll back +8
        int x  = (wx * 16 + qx + 8) & 255;
        size_t di = ((size_t)((b * 256 + y) * 256 + x)) * 192 + h * 32 + n * 16 + nlo;
        obuf[di] = (__bf16)(o[r] * rs[r]);
      }
    }
  }
}

// ---------------------------------------------------------------------------
// Kernel 4: output projection (bf16 GEMM), fp32 result to d_out.
// ---------------------------------------------------------------------------
__global__ __launch_bounds__(256) void proj_o_kernel(const __bf16* __restrict__ obuf,
                                                     const __bf16* __restrict__ woT,
                                                     const float* __restrict__ bo,
                                                     float* __restrict__ out) {
  const int lane  = threadIdx.x & 31;
  const int wave  = threadIdx.x >> 5;
  const int koff  = (lane < 16) ? 0 : 8;
  const int nlo   = lane & 15;
  const int rhalf = (lane >> 4) << 3;
  const int rowbase = blockIdx.x * 128 + wave * 16;
  const int rowA = rowbase + nlo;

  v8f acc[12] = {};
#pragma unroll
  for (int kk = 0; kk < 6; ++kk) {
    v16bf af = ldfrag_bf16(obuf + (size_t)rowA * 192 + kk * 32, koff);
#pragma unroll
    for (int ct = 0; ct < 12; ++ct) {
      v16bf bf = ldfrag_bf16(woT + (size_t)(ct * 16 + nlo) * 192 + kk * 32, koff);
      acc[ct] = wmma_bf16(af, bf, acc[ct]);
    }
  }
#pragma unroll
  for (int ct = 0; ct < 12; ++ct) {
    const int c = ct * 16 + nlo;
    const float bias = bo[c];
#pragma unroll
    for (int r = 0; r < 8; ++r) {
      int p = rowbase + r + rhalf;
      out[(size_t)p * 192 + c] = acc[ct][r] + bias;
    }
  }
}

// ---------------------------------------------------------------------------
extern "C" void kernel_launch(void* const* d_in, const int* in_sizes, int n_in,
                              void* d_out, int out_size, void* d_ws, size_t ws_size,
                              hipStream_t stream) {
  const float* x    = (const float*)d_in[0];
  const float* z    = (const float*)d_in[1];
  const unsigned char* mask = (const unsigned char*)d_in[2];
  const float* Wq   = (const float*)d_in[3];
  const float* bq   = (const float*)d_in[4];
  const float* Wkv  = (const float*)d_in[5];
  const float* bkv  = (const float*)d_in[6];
  const float* Wo   = (const float*)d_in[7];
  const float* bo   = (const float*)d_in[8];
  const float* rel  = (const float*)d_in[9];
  float* out = (float*)d_out;

  __bf16* ws   = (__bf16*)d_ws;
  __bf16* qbuf = ws + OFF_Q;
  __bf16* kbuf = ws + OFF_K;
  __bf16* vbuf = ws + OFF_V;
  __bf16* obuf = ws + OFF_O;
  __bf16* wqT  = ws + OFF_WQT;
  __bf16* wkvT = ws + OFF_WKVT;
  __bf16* woT  = ws + OFF_WOT;

  prep_weights<<<576, 256, 0, stream>>>(Wq, Wkv, Wo, wqT, wkvT, woT);
  proj_q_kernel<<<2048, 256, 0, stream>>>(x, wqT, bq, qbuf);
  proj_kv_kernel<<<1024, 256, 0, stream>>>(z, wkvT, bkv, kbuf, vbuf);
  attn_kernel<<<6144, 256, 0, stream>>>(qbuf, kbuf, vbuf, mask, rel, obuf);
  proj_o_kernel<<<2048, 256, 0, stream>>>(obuf, woT, bo, out);
}
